// WaveNet_35673998360802
// MI455X (gfx1250) — matrix operand order repaired and checked by reading, not compiled, not run on hardware
//
#include <hip/hip_runtime.h>

typedef unsigned int   u32;
typedef unsigned short u16;

typedef __attribute__((ext_vector_type(16))) __bf16 v16bf;
typedef __attribute__((ext_vector_type(8)))  float  v8f;

union BFrag {
    v16bf v;
    uint4 q[2];
};

#define RCH    128
#define TLEN   16384
#define BATCH  8
#define NLAYER 9
#define NT     128   // timesteps per WG (8 waves x 16)
#define LSTR   136   // padded LDS row stride (bf16 elements)

__device__ __forceinline__ u16 f2bf(float f) {
    u32 u = __float_as_uint(f);
    u32 r = (u + 0x7fffu + ((u >> 16) & 1u)) >> 16;   // round-to-nearest-even
    return (u16)r;
}
__device__ __forceinline__ float bf2f(u32 h) {
    return __uint_as_float(h << 16);
}

// Fast activations: one v_exp_f32 + one raw v_rcp_f32 each (no IEEE div chain).
__device__ __forceinline__ float fast_sigm(float x) {
    return __builtin_amdgcn_rcpf(1.0f + __expf(-x));
}
__device__ __forceinline__ float fast_tanh(float x) {
    float ax = fabsf(x);
    float e  = __expf(-2.0f * ax);
    float r  = (1.0f - e) * __builtin_amdgcn_rcpf(1.0f + e);
    return copysignf(r, x);
}

// ---------------------------------------------------------------------------
// Weight prep: fp32 -> bf16 in WMMA A-fragment order.
// A-fragment (16x32 bf16, M x K): lane l (0..31), element e (0..15):
//   oc = mt*16 + (l & 15), hf = l >> 4
//   ic = kt*32 + (e < 8 ? hf*8 + e : 16 + hf*8 + (e - 8))
// ---------------------------------------------------------------------------
__global__ void k_prep_conv(const float* __restrict__ w, u16* __restrict__ dst) {
    int i = blockIdx.x * 256 + threadIdx.x;       // < 9 * 98304
    int l = i / 98304;
    int r = i % 98304;
    int e    = r & 15;  int t1 = r >> 4;
    int lane = t1 & 31; t1 >>= 5;
    int kt   = t1 & 3;  t1 >>= 2;
    int mt   = t1 & 15; t1 >>= 4;
    int tap  = t1;                                 // 0..2
    int oc = mt * 16 + (lane & 15);
    int hf = lane >> 4;
    int ic = kt * 32 + (e < 8 ? hf * 8 + e : 16 + hf * 8 + (e - 8));
    // conv_w layout: [L][2R][R][K]
    float v = w[(((size_t)l * 256 + oc) * 128 + ic) * 3 + tap];
    dst[i] = f2bf(v);
}

__global__ void k_prep_out(const float* __restrict__ w, u16* __restrict__ dst) {
    int i = blockIdx.x * 256 + threadIdx.x;       // < 9 * 16384
    int l = i / 16384;
    int r = i % 16384;
    int e    = r & 15;  int t1 = r >> 4;
    int lane = t1 & 31; t1 >>= 5;
    int kt   = t1 & 3;  t1 >>= 2;
    int mt   = t1;                                 // 0..7
    int oc = mt * 16 + (lane & 15);
    int hf = lane >> 4;
    int ic = kt * 32 + (e < 8 ? hf * 8 + e : 16 + hf * 8 + (e - 8));
    // out_w layout: [L][R][R][1]
    float v = w[((size_t)l * 128 + oc) * 128 + ic];
    dst[i] = f2bf(v);
}

__global__ void k_prep_w1(const float* __restrict__ w, u16* __restrict__ dst) {
    int i = blockIdx.x * 256 + threadIdx.x;       // < 9 * 16384
    int l = i / 16384;
    int r = i % 16384;
    int e    = r & 15;  int t1 = r >> 4;
    int lane = t1 & 31; t1 >>= 5;
    int kt   = t1 & 3;  t1 >>= 2;
    int mt   = t1;                                 // 0..7
    int oc = mt * 16 + (lane & 15);
    int hf = lane >> 4;
    int ic = kt * 32 + (e < 8 ? hf * 8 + e : 16 + hf * 8 + (e - 8));
    // output1_w layout: [R][R*L][1]
    float v = w[(size_t)oc * (128 * NLAYER) + l * 128 + ic];
    dst[i] = f2bf(v);
}

// ---------------------------------------------------------------------------
// Input layer: h0[b][t][c] = bf16(tanh(iw[c]*x[b][t] + ib[c])); skip = 0.
// ---------------------------------------------------------------------------
__global__ void k_input(const float* __restrict__ x, const float* __restrict__ iw,
                        const float* __restrict__ ib,
                        u16* __restrict__ h, float* __restrict__ skip) {
    size_t i = (size_t)blockIdx.x * 256 + threadIdx.x;   // < B*T*R
    int c = (int)(i % RCH);
    size_t bt = i / RCH;
    float v = fast_tanh(iw[c] * x[bt] + ib[c]);
    h[i] = f2bf(v);
    skip[i] = 0.0f;
}

// ---------------------------------------------------------------------------
// One WaveNet layer, fully fused:
//   async-stage weights to LDS -> dilated conv (WMMA, A-frags from LDS) ->
//   gated activation -> LDS transpose -> res + skip GEMMs (WMMA) ->
//   residual add + skip RMW.
// Grid: (TLEN/NT, BATCH), block 256 (8 waves; wave owns 16 timesteps).
// ---------------------------------------------------------------------------
__global__ __launch_bounds__(256) void k_layer(
    const u16* __restrict__ hin, u16* __restrict__ hout,
    float* __restrict__ skip,
    const u16* __restrict__ convA, const u16* __restrict__ outA,
    const u16* __restrict__ w1A,
    const float* __restrict__ cb, const float* __restrict__ ob,
    int dil)
{
    __shared__ u16 sConv[98304];   // 192 KB: conv A-fragments (3 taps)
    __shared__ u16 sOut [8192];    //  16 KB: res-projection A-fragments
    __shared__ u16 sW1  [8192];    //  16 KB: skip-projection A-fragments
    __shared__ u16 sZ   [LSTR * NT]; // 34 KB: z transpose tile

    const int tid  = threadIdx.x;
    const int wv   = tid >> 5;
    const int lane = tid & 31;
    const int hi   = lane >> 4;
    const int l16  = lane & 15;
    const int b    = blockIdx.y;
    const size_t bbase = (size_t)b * TLEN;

    // ---- async-stage all weight fragments into LDS (ASYNCcnt tracked)
    {
        const u32 ldsConv = (u32)(uintptr_t)(&sConv[0]);
        const u32 ldsOut  = (u32)(uintptr_t)(&sOut[0]);
        const u32 ldsW1   = (u32)(uintptr_t)(&sW1[0]);
#pragma unroll 1
        for (int i = tid; i < 12288; i += 256) {       // 12288 x 16B = 192 KB
            u32 off = (u32)i * 16u;
            asm volatile("global_load_async_to_lds_b128 %0, %1, %2"
                         :: "v"(ldsConv + off), "v"(off), "s"(convA) : "memory");
        }
        u32 off = (u32)tid * 16u;
#pragma unroll
        for (int k = 0; k < 4; ++k) {                  // 1024 x 16B each
            asm volatile("global_load_async_to_lds_b128 %0, %1, %2"
                         :: "v"(ldsOut + off), "v"(off), "s"(outA) : "memory");
            asm volatile("global_load_async_to_lds_b128 %0, %1, %2"
                         :: "v"(ldsW1 + off), "v"(off), "s"(w1A) : "memory");
            off += 256u * 16u;
        }
        asm volatile("s_wait_asynccnt 0" ::: "memory");
        __syncthreads();
    }

    const int tw = blockIdx.x * NT + wv * 16;

    // ---- dilated conv: pre[256,16] = sum_{tap,kt} W_tap[256,128]*h[.,t-off]
    v8f acc[16];
#pragma unroll
    for (int m = 0; m < 16; ++m) acc[m] = (v8f){0.f,0.f,0.f,0.f,0.f,0.f,0.f,0.f};

#pragma unroll
    for (int kt = 0; kt < 4; ++kt) {
#pragma unroll
        for (int tap = 0; tap < 3; ++tap) {
            int tt = tw + l16 + (tap - 2) * dil;       // causal taps: -2d,-d,t
            int ts = tt < 0 ? 0 : tt;
            const uint4* p = (const uint4*)(hin + (bbase + ts) * RCH + kt * 32 + hi * 8);
            BFrag bf;
            uint4 q0 = p[0], q1 = p[2];
            if (tt < 0) {                              // left pad -> zeros
                q0 = (uint4){0u,0u,0u,0u};
                q1 = (uint4){0u,0u,0u,0u};
            }
            bf.q[0] = q0; bf.q[1] = q1;
#pragma unroll
            for (int mt = 0; mt < 16; ++mt) {
                BFrag af;
                const uint4* ap = (const uint4*)(sConv +
                    ((((tap * 16 + mt) * 4 + kt) * 32 + lane) * 16));
                af.q[0] = ap[0]; af.q[1] = ap[1];
                acc[mt] = __builtin_amdgcn_wmma_f32_16x16x32_bf16(
                    false, af.v, false, bf.v, (short)0, acc[mt], false, false);
            }
        }
    }

    // ---- gated activation z = tanh(t+bt)*sigmoid(g+bg); z -> LDS [t][c]
    const int zrow = wv * 16 + l16;
#pragma unroll
    for (int mt = 0; mt < 8; ++mt) {
#pragma unroll
        for (int j = 0; j < 8; j += 2) {
            int c = mt * 16 + hi * 8 + j;
            float t0 = acc[mt][j]     + cb[c];
            float t1 = acc[mt][j + 1] + cb[c + 1];
            float g0 = acc[mt + 8][j]     + cb[RCH + c];
            float g1 = acc[mt + 8][j + 1] + cb[RCH + c + 1];
            float z0 = fast_tanh(t0) * fast_sigm(g0);
            float z1 = fast_tanh(t1) * fast_sigm(g1);
            u32 pk = (u32)f2bf(z0) | ((u32)f2bf(z1) << 16);
            *(u32*)&sZ[zrow * LSTR + c] = pk;          // 4B aligned (row,c even)
        }
    }

    // ---- res = out_w @ z ; skipacc = w1_l @ z   (both 128x128 GEMMs)
    v8f racc[8], sacc[8];
#pragma unroll
    for (int m = 0; m < 8; ++m) {
        racc[m] = (v8f){0.f,0.f,0.f,0.f,0.f,0.f,0.f,0.f};
        sacc[m] = (v8f){0.f,0.f,0.f,0.f,0.f,0.f,0.f,0.f};
    }
#pragma unroll
    for (int kt = 0; kt < 4; ++kt) {
        BFrag zf;
        const uint4* zp = (const uint4*)&sZ[(wv * 16 + l16) * LSTR + kt * 32 + hi * 8];
        zf.q[0] = zp[0]; zf.q[1] = zp[2];
#pragma unroll
        for (int mt = 0; mt < 8; ++mt) {
            BFrag ar;
            const uint4* rp = (const uint4*)(sOut + (((mt * 4 + kt) * 32 + lane) * 16));
            ar.q[0] = rp[0]; ar.q[1] = rp[1];
            racc[mt] = __builtin_amdgcn_wmma_f32_16x16x32_bf16(
                false, ar.v, false, zf.v, (short)0, racc[mt], false, false);
            BFrag a1;
            const uint4* wp = (const uint4*)(sW1 + (((mt * 4 + kt) * 32 + lane) * 16));
            a1.q[0] = wp[0]; a1.q[1] = wp[1];
            sacc[mt] = __builtin_amdgcn_wmma_f32_16x16x32_bf16(
                false, a1.v, false, zf.v, (short)0, sacc[mt], false, false);
        }
    }

    // ---- h_{i+1} = h_i + res + ob ; skip += sacc
    const int t = tw + l16;
#pragma unroll
    for (int mt = 0; mt < 8; ++mt) {
#pragma unroll
        for (int j = 0; j < 8; j += 2) {
            int c = mt * 16 + hi * 8 + j;
            size_t idx = (bbase + t) * RCH + c;
            u32 hold = *(const u32*)(hin + idx);
            float h0 = bf2f(hold & 0xffffu) + racc[mt][j]     + ob[c];
            float h1 = bf2f(hold >> 16)     + racc[mt][j + 1] + ob[c + 1];
            *(u32*)(hout + idx) = (u32)f2bf(h0) | ((u32)f2bf(h1) << 16);
            float2 sv = *(float2*)(skip + idx);
            sv.x += sacc[mt][j];
            sv.y += sacc[mt][j + 1];
            *(float2*)(skip + idx) = sv;
        }
    }
}

// ---------------------------------------------------------------------------
// Output head: out[b][t] = b2 + sum_c w2[c] * tanh(skip[b][t][c] + b1[c])
// ---------------------------------------------------------------------------
__global__ void k_out(const float* __restrict__ skip, const float* __restrict__ b1,
                      const float* __restrict__ w2, const float* __restrict__ b2,
                      float* __restrict__ out) {
    size_t bt = (size_t)blockIdx.x * 256 + threadIdx.x;  // < B*T
    const float* sp = skip + bt * RCH;
    float s = 0.0f;
#pragma unroll 8
    for (int c = 0; c < RCH; ++c)
        s += w2[c] * fast_tanh(sp[c] + b1[c]);
    out[bt] = s + b2[0];
}

// ---------------------------------------------------------------------------
extern "C" void kernel_launch(void* const* d_in, const int* in_sizes, int n_in,
                              void* d_out, int out_size, void* d_ws, size_t ws_size,
                              hipStream_t stream) {
    const float* x    = (const float*)d_in[0];
    const float* iw   = (const float*)d_in[1];
    const float* ib   = (const float*)d_in[2];
    const float* cw   = (const float*)d_in[3];
    const float* cb   = (const float*)d_in[4];
    const float* ow   = (const float*)d_in[5];
    const float* ob   = (const float*)d_in[6];
    const float* w1   = (const float*)d_in[7];
    const float* b1   = (const float*)d_in[8];
    const float* w2   = (const float*)d_in[9];
    const float* b2   = (const float*)d_in[10];

    const size_t HSZ = (size_t)BATCH * TLEN * RCH;          // activation elements
    char* ws = (char*)d_ws;
    u16*   hA    = (u16*)(ws);                              // bf16, 32 MB
    u16*   hB    = (u16*)(ws + HSZ * 2);                    // bf16, 32 MB
    float* skip  = (float*)(ws + HSZ * 4);                  // fp32, 64 MB
    u16*   convA = (u16*)(ws + HSZ * 8);                    // 9*98304 bf16
    u16*   outA  = (u16*)(ws + HSZ * 8 + (size_t)NLAYER * 98304 * 2);
    u16*   w1A   = (u16*)(ws + HSZ * 8 + (size_t)NLAYER * 98304 * 2
                                       + (size_t)NLAYER * 16384 * 2);

    // Weight shuffles into WMMA fragment order (bf16)
    k_prep_conv<<<(NLAYER * 98304) / 256, 256, 0, stream>>>(cw, convA);
    k_prep_out <<<(NLAYER * 16384) / 256, 256, 0, stream>>>(ow, outA);
    k_prep_w1  <<<(NLAYER * 16384) / 256, 256, 0, stream>>>(w1, w1A);

    // Input 1x1 conv + tanh, zero the skip accumulator
    k_input<<<(unsigned)(HSZ / 256), 256, 0, stream>>>(x, iw, ib, hA, skip);

    static const int DIL[NLAYER] = {1, 2, 4, 8, 16, 32, 64, 128, 256};
    u16* hin = hA;
    u16* hot = hB;
    for (int l = 0; l < NLAYER; ++l) {
        k_layer<<<dim3(TLEN / NT, BATCH), 256, 0, stream>>>(
            hin, hot, skip,
            convA + (size_t)l * 98304,
            outA  + (size_t)l * 16384,
            w1A   + (size_t)l * 16384,
            cb + (size_t)l * 256,
            ob + (size_t)l * 128,
            DIL[l]);
        u16* tmp = hin; hin = hot; hot = tmp;
    }

    // Output head
    k_out<<<(BATCH * TLEN) / 256, 256, 0, stream>>>(skip, b1, w2, b2, (float*)d_out);
}
